// EpisodicThemeMemory_52518860095973
// MI455X (gfx1250) — matrix operand order repaired and checked by reading, not compiled, not run on hardware
//
#include <hip/hip_runtime.h>
#include <hip/hip_bf16.h>

// ---------------------------------------------------------------------------
// EpisodicThemeMemory fused kernels for gfx1250 (MI455X, wave32, WMMA bf16,
// TDM tensor_load_to_lds with LDS padding for bank-conflict-free fragments)
// ---------------------------------------------------------------------------
#define B_  32
#define N_  4096
#define D_  256
#define M_  64
#define H_  4
#define HD_ 64
#define D2_ 128
#define TOK_TILE 64
#define THREADS  256       // 8 waves of 32

// padded LDS strides (elements); all row strides are multiples of 16 bytes
#define LDA   264          // 256-wide bf16 mats: 528 B rows -> 4-bank stagger
#define LDV   72           // V^T (64 wide):      144 B rows
#define LDATT 88           // attn (64 wide):     176 B rows
#define LDSF  68           // scores f32 (64 wide): 272 B rows
#define LDH   136          // wg hidden (128 wide): 272 B rows

typedef __attribute__((ext_vector_type(16))) __bf16 v16bf;
typedef __attribute__((ext_vector_type(8)))  float  v8f;
typedef unsigned int u32x4 __attribute__((ext_vector_type(4)));
typedef int          i32x4 __attribute__((ext_vector_type(4)));
typedef int          i32x8 __attribute__((ext_vector_type(8)));

struct U128x2 { uint4 lo; uint4 hi; };   // 32 bytes == v16bf

__device__ __forceinline__ unsigned short f2bf(float f) {
    unsigned u = __builtin_bit_cast(unsigned, f);
    unsigned r = u + 0x7FFFu + ((u >> 16) & 1u);   // round-nearest-even
    return (unsigned short)(r >> 16);
}
__device__ __forceinline__ float bf2f(unsigned short h) {
    unsigned u = ((unsigned)h) << 16;
    return __builtin_bit_cast(float, u);
}

// ---------------------------------------------------------------------------
// TDM: 1-D tile load global->LDS, optional LDS padding (descriptor per ISA §8)
//   data_size = 8B units; pad_amount/interval encode the padded row strides.
// ---------------------------------------------------------------------------
#if __has_builtin(__builtin_amdgcn_tensor_load_to_lds)
#define HAVE_TDM 1
#else
#define HAVE_TDM 0
#endif

__device__ __forceinline__ void tdm_wait0() {
#if __has_builtin(__builtin_amdgcn_s_wait_tensorcnt)
    __builtin_amdgcn_s_wait_tensorcnt(0);
#else
    asm volatile("s_wait_tensorcnt 0" ::: "memory");
#endif
}

#if HAVE_TDM
__device__ __forceinline__ void tdm_load_1d(void* lds_ptr, const void* gptr,
                                            unsigned units8, bool pad_en,
                                            unsigned pad_interval,  // 1<<i 8B-units
                                            unsigned pad_amount) {  // dwords-1
    unsigned long long ga = (unsigned long long)(size_t)gptr;
    unsigned lds_addr = (unsigned)(size_t)lds_ptr;   // flat LDS addr[31:0]
    u32x4 g0;
    g0[0] = 1u;                                       // count=1 (valid), no gather
    g0[1] = lds_addr;
    g0[2] = (unsigned)ga;                             // global_addr[31:0]
    g0[3] = ((unsigned)(ga >> 32) & 0x01FFFFFFu) | (2u << 30);  // addr[56:32],type=2
    i32x8 g1;
    unsigned w0 = (3u << 16)                           // data_size = 8 bytes
                | (pad_en ? (1u << 20) : 0u)
                | ((pad_interval & 7u) << 22)
                | ((pad_amount & 127u) << 25);
    g1[0] = (int)w0;
    g1[1] = (int)(0xFFFFu << 16);                      // tensor_dim0 lo16 (huge)
    g1[2] = (int)0x7FFFu;                              // tensor_dim0 hi16; dim1=0
    g1[3] = (int)((units8 & 0xFFFFu) << 16);           // tile_dim0
    g1[4] = 0;                                         // tile_dim1/2 = 0 (1-D)
    g1[5] = 0; g1[6] = 0; g1[7] = 0;                   // strides unused (1-D)
    i32x4 z4 = {0, 0, 0, 0};
#if defined(__clang_major__) && (__clang_major__ >= 23)
    i32x8 z8 = {0, 0, 0, 0, 0, 0, 0, 0};
    __builtin_amdgcn_tensor_load_to_lds(g0, g1, z4, z4, z8, 0);
#else
    __builtin_amdgcn_tensor_load_to_lds(g0, g1, z4, z4, 0);
#endif
}
#endif

// ---------------------------------------------------------------------------
// WMMA tile helper: C(16x16 f32) = A(16xK bf16, LDS) @ W^T, W row-major (NxK)
// A layout (ISA 7.12.2): lane%16 = M; lanes<16: e0-7 -> k0+0..7, e8-15 -> k0+16..23
//                                    lanes>=16: e0-7 -> k0+8..15, e8-15 -> k0+24..31
// B layout (Kx16): lane%16 = N; same k mapping as A (e0-7 -> k0+8h+0..7, e8-15 -> k0+16+8h+0..7)
// C layout: VGPR r, lane L: M = r + (L>=16 ? 8 : 0), N = L%16
// ---------------------------------------------------------------------------
template <int K>
__device__ __forceinline__ v8f wmma_tile_AxWT(const unsigned short* __restrict__ A, int lda,
                                              int rowBase,
                                              const unsigned short* __restrict__ W, int ldw,
                                              int colBase, float bias) {
    const int lane = threadIdx.x & 31;
    const bool hi  = lane >= 16;
    const int ln   = lane & 15;
    v8f c;
#pragma unroll
    for (int i = 0; i < 8; ++i) c[i] = bias;
    const unsigned short* pa0 = A + (size_t)(rowBase + ln) * lda + (hi ? 8 : 0);
    const unsigned short* pb0 = W + (size_t)(colBase + ln) * ldw + (hi ? 8 : 0);
#pragma unroll
    for (int k0 = 0; k0 < K; k0 += 32) {
        const unsigned short* pa = pa0 + k0;
        const unsigned short* pb = pb0 + k0;
        v16bf a = __builtin_bit_cast(v16bf,
            U128x2{ *(const uint4*)(pa), *(const uint4*)(pa + 16) });
        v16bf b = __builtin_bit_cast(v16bf,
            U128x2{ *(const uint4*)(pb), *(const uint4*)(pb + 16) });
        c = __builtin_amdgcn_wmma_f32_16x16x32_bf16(false, a, false, b,
                                                    (short)0, c, false, false);
    }
    return c;
}

__device__ __forceinline__ void store_tile_bf16(unsigned short* __restrict__ Out, int ldo,
                                                int rowBase, int colBase, const v8f& c,
                                                bool relu) {
    const int lane = threadIdx.x & 31;
    const bool hi  = lane >= 16;
    const int ln   = lane & 15;
#pragma unroll
    for (int r = 0; r < 8; ++r) {
        float v = c[r];
        if (relu && v < 0.f) v = 0.f;
        Out[(size_t)(rowBase + r + (hi ? 8 : 0)) * ldo + colBase + ln] = f2bf(v);
    }
}

// ---------------------------------------------------------------------------
// Kernel 0: fp32 -> bf16 weight conversion
// ---------------------------------------------------------------------------
__global__ void k_cvt(const float* __restrict__ src, unsigned short* __restrict__ dst, int n) {
    int i = blockIdx.x * blockDim.x + threadIdx.x;
    if (i < n) dst[i] = f2bf(src[i]);
}

// ---------------------------------------------------------------------------
// Kernel 1: K/V projection of memory. K stored (b, m, d) row-major;
//           V stored TRANSPOSED (b, d, m) so ctx GEMM's B operand is K-contig.
// ---------------------------------------------------------------------------
__global__ __launch_bounds__(THREADS) void k_kv(const float* __restrict__ memory,
                                                const float* __restrict__ in_w,
                                                const float* __restrict__ in_b,
                                                unsigned short* __restrict__ Kb,
                                                unsigned short* __restrict__ Vt) {
    __shared__ float mrow[D_];
    const int bm  = blockIdx.x;        // b*M + m
    const int b   = bm >> 6;
    const int m   = bm & (M_ - 1);
    const int tid = threadIdx.x;
    mrow[tid] = memory[(size_t)bm * D_ + tid];
    __syncthreads();
    const int j = tid;                 // output feature
    float accK = in_b[D_ + j];
    float accV = in_b[2 * D_ + j];
    const float* wkr = in_w + (size_t)(D_ + j) * D_;
    const float* wvr = in_w + (size_t)(2 * D_ + j) * D_;
#pragma unroll 4
    for (int i = 0; i < D_; ++i) {
        float x = mrow[i];
        accK += x * wkr[i];
        accV += x * wvr[i];
    }
    Kb[(size_t)bm * D_ + j]             = f2bf(accK);
    Vt[((size_t)b * D_ + j) * M_ + m]   = f2bf(accV);
}

// ---------------------------------------------------------------------------
// Kernel 2: fused main path. One WG = 64 tokens x one batch, 8 waves.
// ---------------------------------------------------------------------------
#define SM_XF   0
#define SM_A    (SM_XF  + TOK_TILE * D_ * 4)       //  65536 (x / residual, f32)
#define SM_B    (SM_A   + TOK_TILE * LDA * 2)      // +33792 (x_bf -> ln_x -> qq)
#define SM_K    (SM_B   + TOK_TILE * LDA * 2)      // +33792 (q)
#define SM_V    (SM_K   + TOK_TILE * LDA * 2)      // +33792 (K, padded)
#define SM_C    (SM_V   + D_ * LDV * 2)            // +36864 (V^T, padded)
#define SM_SCR  (SM_C   + TOK_TILE * LDA * 2)      // +33792 (ctx)
#define SM_ATT  (SM_SCR + TOK_TILE * LDSF * 4)     // +17408 (wg hidden / scores)
#define SM_ST   (SM_ATT + TOK_TILE * LDATT * 2)    // +11264 (attn bf16)
#define SM_TOTAL (SM_ST + 2 * TOK_TILE * 4)        //   +512  => 266752 B

__global__ __launch_bounds__(THREADS, 1)
void k_main(const float* __restrict__ phrases,
            const float* __restrict__ q_b,  const float* __restrict__ in_b,
            const float* __restrict__ out_b,
            const float* __restrict__ pre_g, const float* __restrict__ pre_b,
            const float* __restrict__ ln_g,  const float* __restrict__ ln_b,
            const float* __restrict__ wg_b1, const float* __restrict__ wg_w2,
            const float* __restrict__ wg_b2,
            const unsigned short* __restrict__ qw_bf,
            const unsigned short* __restrict__ wq_bf,
            const unsigned short* __restrict__ ow_bf,
            const unsigned short* __restrict__ wg1_bf,
            const unsigned short* __restrict__ Kbf,
            const unsigned short* __restrict__ Vtf,
            float* __restrict__ scores_out,
            float* __restrict__ enhanced) {
    extern __shared__ __align__(16) unsigned char smem[];
    float*          xf    = (float*)(smem + SM_XF);
    unsigned short* Abuf  = (unsigned short*)(smem + SM_A);
    unsigned short* Bbuf  = (unsigned short*)(smem + SM_B);
    unsigned short* Kbuf  = (unsigned short*)(smem + SM_K);
    unsigned short* Vbuf  = (unsigned short*)(smem + SM_V);   // V^T: 256 x 64 (ld LDV)
    unsigned short* Cbuf  = (unsigned short*)(smem + SM_C);
    unsigned short* Hbuf  = (unsigned short*)(smem + SM_SCR); // 64 x 128 (ld LDH)
    float*          Sf    = (float*)(smem + SM_SCR);          // 64 x 64  (ld LDSF)
    unsigned short* attnb = (unsigned short*)(smem + SM_ATT);
    float*          muS   = (float*)(smem + SM_ST);
    float*          rsS   = (float*)(smem + SM_ST) + TOK_TILE;

    const int tid  = threadIdx.x;
    const int wave = tid >> 5;
    const int lane = tid & 31;
    const bool hi  = lane >= 16;
    const int ln   = lane & 15;

    const int b    = blockIdx.x >> 6;           // 64 tiles per batch
    const int tile = blockIdx.x & 63;
    const size_t base = ((size_t)b * N_ + (size_t)tile * TOK_TILE) * D_;

    // ---- stage 1: x tile + K + V^T into LDS (TDM where available) ----------
#if HAVE_TDM
    if (tid < 32) {
        // x: 65536 B contiguous, no padding
        tdm_load_1d(xf, phrases + base, (64 * 256 * 4) / 8, false, 0, 0);
        // K: 64 rows x 512 B, pad 16 B (4 dwords) every 512 B (64 x 8B units)
        tdm_load_1d(Kbuf, Kbf + (size_t)b * M_ * D_, (64 * 256 * 2) / 8,
                    true, 6, 3);
        // V^T: 256 rows x 128 B, pad 16 B every 128 B (16 x 8B units)
        tdm_load_1d(Vbuf, Vtf + (size_t)b * D_ * M_, (256 * 64 * 2) / 8,
                    true, 4, 3);
        tdm_wait0();
    }
#else
    {
        const float4* src = (const float4*)(phrases + base);
        float4* dst = (float4*)xf;
        for (int i = tid; i < TOK_TILE * D_ / 4; i += THREADS) dst[i] = src[i];
        const uint4* ks = (const uint4*)(Kbf + (size_t)b * M_ * D_);
        for (int i = tid; i < 64 * 32; i += THREADS)
            ((uint4*)(Kbuf + (size_t)(i >> 5) * LDA))[i & 31] = ks[i];
        const uint4* vs = (const uint4*)(Vtf + (size_t)b * D_ * M_);
        for (int i = tid; i < 256 * 8; i += THREADS)
            ((uint4*)(Vbuf + (size_t)(i >> 3) * LDV))[i & 7] = vs[i];
    }
#endif
    if (tid < 128) {   // warm weights into cache (global_prefetch_b8)
        __builtin_prefetch(qw_bf + (size_t)tid * 512, 0, 1);
        __builtin_prefetch(wq_bf + (size_t)tid * 512, 0, 1);
        __builtin_prefetch(ow_bf + (size_t)tid * 512, 0, 1);
    }
    __syncthreads();

    // ---- stage 2: pre-LN stats (bank-staggered) ; stage 3: x -> bf16 ------
    if (tid < TOK_TILE) {
        const float* r = xf + (size_t)tid * D_;
        const int st = (tid << 2);
        float s = 0.f, ss = 0.f;
        for (int i = 0; i < D_; ++i) {
            float v = r[(i + st) & (D_ - 1)];
            s += v; ss += v * v;
        }
        float mu = s * (1.0f / D_);
        float var = ss * (1.0f / D_) - mu * mu;
        muS[tid] = mu;
        rsS[tid] = 1.0f / sqrtf(var + 1e-5f);
    }
    for (int i = tid; i < TOK_TILE * D_; i += THREADS) {
        int row = i >> 8, col = i & (D_ - 1);
        Abuf[(size_t)row * LDA + col] = f2bf(xf[i]);
    }
    __syncthreads();

    // ---- stage 4: write-gate layer 1: H = relu(x @ wg1^T + b1) (64x128) ---
    for (int t = wave; t < 32; t += 8) {
        int tr = t >> 3, tc = t & 7;
        v8f c = wmma_tile_AxWT<D_>(Abuf, LDA, tr * 16, wg1_bf, D_, tc * 16,
                                   wg_b1[tc * 16 + ln]);
        store_tile_bf16(Hbuf, LDH, tr * 16, tc * 16, c, true);
    }
    __syncthreads();

    // ---- stage 5: gate score (sigmoid) ; stage 6: ln_x bf16 -> Abuf -------
    if (tid < TOK_TILE) {
        float s = wg_b2[0];
        const unsigned short* hr = Hbuf + (size_t)tid * LDH;
        for (int j = 0; j < D2_; ++j) s += bf2f(hr[j]) * wg_w2[j];
        s = 1.0f / (1.0f + __expf(-s));
        scores_out[(size_t)b * N_ + (size_t)tile * TOK_TILE + tid] = s;
    }
    for (int i = tid; i < TOK_TILE * D_; i += THREADS) {
        int row = i >> 8, col = i & (D_ - 1);
        Abuf[(size_t)row * LDA + col] =
            f2bf((xf[i] - muS[row]) * rsS[row] * pre_g[col] + pre_b[col]);
    }
    __syncthreads();

    // ---- stage 7: q = ln_x @ q_w^T + q_b  -> Bbuf -------------------------
    for (int t = wave; t < 64; t += 8) {
        int tr = t >> 4, tc = t & 15;
        v8f c = wmma_tile_AxWT<D_>(Abuf, LDA, tr * 16, qw_bf, D_, tc * 16,
                                   q_b[tc * 16 + ln]);
        store_tile_bf16(Bbuf, LDA, tr * 16, tc * 16, c, false);
    }
    __syncthreads();

    // ---- stage 8: qq = q @ wq^T + bq  -> Abuf (ln_x dead) -----------------
    for (int t = wave; t < 64; t += 8) {
        int tr = t >> 4, tc = t & 15;
        v8f c = wmma_tile_AxWT<D_>(Bbuf, LDA, tr * 16, wq_bf, D_, tc * 16,
                                   in_b[tc * 16 + ln]);
        store_tile_bf16(Abuf, LDA, tr * 16, tc * 16, c, false);
    }
    __syncthreads();

    // ---- stage 9: per-head attention (scores -> softmax -> ctx) -----------
    for (int h = 0; h < H_; ++h) {
        // scores(64x64) = qq_h @ K_h^T * (1/8)
        for (int t = wave; t < 16; t += 8) {
            int tr = t >> 2, tc = t & 3;
            v8f c = wmma_tile_AxWT<HD_>(Abuf + h * HD_, LDA, tr * 16,
                                        Kbuf + h * HD_, LDA, tc * 16, 0.0f);
#pragma unroll
            for (int r = 0; r < 8; ++r)
                Sf[(size_t)(tr * 16 + r + (hi ? 8 : 0)) * LDSF + tc * 16 + ln] =
                    c[r] * 0.125f;
        }
        __syncthreads();
        // softmax over M=64 per token row
        if (tid < TOK_TILE) {
            const int row = tid;
            const float* sr = Sf + (size_t)row * LDSF;
            float mx = -1e30f;
            for (int m = 0; m < M_; ++m) mx = fmaxf(mx, sr[m]);
            float sum = 0.f;
            for (int m = 0; m < M_; ++m) sum += __expf(sr[m] - mx);
            float inv = 1.0f / sum;
            unsigned short* ar = attnb + (size_t)row * LDATT;
            for (int m = 0; m < M_; ++m) ar[m] = f2bf(__expf(sr[m] - mx) * inv);
        }
        __syncthreads();
        // ctx_h(64x64) = attn @ V_h  (V^T rows are output features, K-contig)
        for (int t = wave; t < 16; t += 8) {
            int tr = t >> 2, tc = t & 3;
            v8f c = wmma_tile_AxWT<M_>(attnb, LDATT, tr * 16,
                                       Vbuf, LDV, h * HD_ + tc * 16, 0.0f);
            store_tile_bf16(Cbuf, LDA, tr * 16, h * HD_ + tc * 16, c, false);
        }
        __syncthreads();
    }

    // ---- stage 10: out = ctx @ out_w^T + out_b ; y = x + out (in xf) ------
    for (int t = wave; t < 64; t += 8) {
        int tr = t >> 4, tc = t & 15;
        v8f c = wmma_tile_AxWT<D_>(Cbuf, LDA, tr * 16, ow_bf, D_, tc * 16,
                                   out_b[tc * 16 + ln]);
#pragma unroll
        for (int r = 0; r < 8; ++r) {
            int row = tr * 16 + r + (hi ? 8 : 0);
            int col = tc * 16 + ln;
            xf[(size_t)row * D_ + col] += c[r];   // RES_SCALE == 1.0
        }
    }
    __syncthreads();

    // ---- stage 11: final LN stats over y (bank-staggered) -----------------
    if (tid < TOK_TILE) {
        const float* r = xf + (size_t)tid * D_;
        const int st = (tid << 2);
        float s = 0.f, ss = 0.f;
        for (int i = 0; i < D_; ++i) {
            float v = r[(i + st) & (D_ - 1)];
            s += v; ss += v * v;
        }
        float mu = s * (1.0f / D_);
        float var = ss * (1.0f / D_) - mu * mu;
        muS[tid] = mu;
        rsS[tid] = 1.0f / sqrtf(var + 1e-5f);
    }
    __syncthreads();

    // ---- stage 12: normalize + coalesced store ----------------------------
    {
        const int col = tid;  // THREADS == D_
        float g = ln_g[col], bb = ln_b[col];
        for (int row = 0; row < TOK_TILE; ++row) {
            float v = (xf[(size_t)row * D_ + col] - muS[row]) * rsS[row] * g + bb;
            enhanced[base + (size_t)row * D_ + col] = v;
        }
    }
}

// ---------------------------------------------------------------------------
// Kernel 3: top-64 (value desc, index asc == jax.lax.top_k ties), index sort,
// gather with lazy candidate reconstruction (norm_p * mask * score).
// ---------------------------------------------------------------------------
__global__ __launch_bounds__(THREADS) void k_merge(const float* __restrict__ phrases,
                                                   const float* __restrict__ memory,
                                                   const float* __restrict__ wscores,
                                                   float* __restrict__ merged) {
    __shared__ float s[M_ + N_];
    __shared__ float rv[THREADS];
    __shared__ int   ri[THREADS];
    __shared__ int   sel[M_];
    const int b = blockIdx.x;
    const int tid = threadIdx.x;
    const int TOT = M_ + N_;

    for (int i = tid; i < TOT; i += THREADS)
        s[i] = (i < M_) ? 0.5f : wscores[(size_t)b * N_ + (i - M_)];
    __syncthreads();

    for (int k = 0; k < M_; ++k) {
        float bv = -1e30f; int bi = 0x7fffffff;
        for (int i = tid; i < TOT; i += THREADS) {
            float v = s[i];
            if (v > bv || (v == bv && i < bi)) { bv = v; bi = i; }
        }
        rv[tid] = bv; ri[tid] = bi;
        __syncthreads();
        for (int off = THREADS / 2; off > 0; off >>= 1) {
            if (tid < off) {
                float v2 = rv[tid + off]; int i2 = ri[tid + off];
                if (v2 > rv[tid] || (v2 == rv[tid] && i2 < ri[tid])) {
                    rv[tid] = v2; ri[tid] = i2;
                }
            }
            __syncthreads();
        }
        if (tid == 0) { sel[k] = ri[0]; s[ri[0]] = -1e30f; }
        __syncthreads();
    }

    if (tid == 0) {  // insertion sort, 64 elements
        for (int i = 1; i < M_; ++i) {
            int key = sel[i], j = i - 1;
            while (j >= 0 && sel[j] > key) { sel[j + 1] = sel[j]; --j; }
            sel[j + 1] = key;
        }
    }
    __syncthreads();

    for (int r = 0; r < M_; ++r) {
        int idx = sel[r];                           // uniform across block
        float* dst = merged + ((size_t)b * M_ + r) * D_;
        if (idx < M_) {
            dst[tid] = memory[((size_t)b * M_ + idx) * D_ + tid];
        } else {
            int tok = idx - M_;
            const float* src = phrases + ((size_t)b * N_ + tok) * D_;
            float v = src[tid];
            rv[tid] = v * v;
            __syncthreads();
            for (int off = THREADS / 2; off > 0; off >>= 1) {
                if (tid < off) rv[tid] += rv[tid + off];
                __syncthreads();
            }
            float nrm = fmaxf(sqrtf(rv[0]), 1e-12f);
            float sc = wscores[(size_t)b * N_ + tok];
            float gate = (sc > 0.5f) ? sc : 0.0f;   // mask * score
            dst[tid] = v / nrm * gate;
            __syncthreads();
        }
    }
}

// ---------------------------------------------------------------------------
// Host launcher
// ---------------------------------------------------------------------------
extern "C" void kernel_launch(void* const* d_in, const int* in_sizes, int n_in,
                              void* d_out, int out_size, void* d_ws, size_t ws_size,
                              hipStream_t stream) {
    const float* phrases = (const float*)d_in[0];
    const float* memory  = (const float*)d_in[1];
    const float* q_w     = (const float*)d_in[2];
    const float* q_b     = (const float*)d_in[3];
    const float* in_w    = (const float*)d_in[4];
    const float* in_b    = (const float*)d_in[5];
    const float* out_w   = (const float*)d_in[6];
    const float* out_b   = (const float*)d_in[7];
    const float* pre_g   = (const float*)d_in[8];
    const float* pre_b   = (const float*)d_in[9];
    const float* ln_g    = (const float*)d_in[10];
    const float* ln_b    = (const float*)d_in[11];
    const float* wg_w1   = (const float*)d_in[12];
    const float* wg_b1   = (const float*)d_in[13];
    const float* wg_w2   = (const float*)d_in[14];
    const float* wg_b2   = (const float*)d_in[15];

    size_t off = 0;
    auto carve = [&](size_t bytes) -> void* {
        void* p = (char*)d_ws + off;
        off += (bytes + 255) & ~(size_t)255;
        return p;
    };
    unsigned short* qw_bf  = (unsigned short*)carve((size_t)D_ * D_ * 2);
    unsigned short* wq_bf  = (unsigned short*)carve((size_t)D_ * D_ * 2);
    unsigned short* ow_bf  = (unsigned short*)carve((size_t)D_ * D_ * 2);
    unsigned short* wg1_bf = (unsigned short*)carve((size_t)D2_ * D_ * 2);
    unsigned short* Kb     = (unsigned short*)carve((size_t)B_ * M_ * D_ * 2);
    unsigned short* Vt     = (unsigned short*)carve((size_t)B_ * D_ * M_ * 2);
    float*          wsc    = (float*)carve((size_t)B_ * N_ * 4);

    k_cvt<<<(D_ * D_ + 255) / 256, 256, 0, stream>>>(q_w, qw_bf, D_ * D_);
    k_cvt<<<(D_ * D_ + 255) / 256, 256, 0, stream>>>(in_w, wq_bf, D_ * D_);
    k_cvt<<<(D_ * D_ + 255) / 256, 256, 0, stream>>>(out_w, ow_bf, D_ * D_);
    k_cvt<<<(D2_ * D_ + 255) / 256, 256, 0, stream>>>(wg_w1, wg1_bf, D2_ * D_);

    k_kv<<<B_ * M_, THREADS, 0, stream>>>(memory, in_w, in_b, Kb, Vt);

    (void)hipFuncSetAttribute((const void*)k_main,
                              hipFuncAttributeMaxDynamicSharedMemorySize,
                              (int)SM_TOTAL);
    float* enhanced = (float*)d_out;
    k_main<<<B_ * (N_ / TOK_TILE), THREADS, SM_TOTAL, stream>>>(
        phrases, q_b, in_b, out_b, pre_g, pre_b, ln_g, ln_b,
        wg_b1, wg_w2, wg_b2, qw_bf, wq_bf, ow_bf, wg1_bf, Kb, Vt,
        wsc, enhanced);

    float* merged = (float*)d_out + (size_t)B_ * N_ * D_;
    k_merge<<<B_, THREADS, 0, stream>>>(phrases, memory, wsc, merged);

    (void)in_sizes; (void)n_in; (void)out_size; (void)ws_size;
}
